// QuantLinearW4Grouped_20675972563264
// MI455X (gfx1250) — matrix operand order repaired and checked by reading, not compiled, not run on hardware
//
#include <hip/hip_runtime.h>
#include <stdint.h>

typedef __attribute__((ext_vector_type(16))) _Float16 v16h;
typedef __attribute__((ext_vector_type(8)))  _Float16 v8h;
typedef __attribute__((ext_vector_type(8)))  float    v8f;

#define M_DIM 2048
#define K_DIM 4096
#define N_DIM 4096

// WGP-scope prefetch (scope field 0 = WGP: fills all cache levels incl. L0).
// __builtin_prefetch can only produce SE/DEV/SYS scope, which merely warms GL2
// -- useless here since the working set already lives in the 192MB L2.
__device__ __forceinline__ void prefetch_wgp(const void* p) {
    asm volatile("global_prefetch_b8 %0, off" :: "v"(p));
}

// ---------------- conversion kernels ----------------

// x: f32 -> f16, 8 elements per thread
__global__ __launch_bounds__(256) void cvt_x_f16(const float* __restrict__ x,
                                                 _Float16* __restrict__ xh) {
    const int i = (blockIdx.x * 256 + threadIdx.x) * 8;
    float4 a = *(const float4*)(x + i);
    float4 b = *(const float4*)(x + i + 4);
    v8h h;
    h[0] = (_Float16)a.x; h[1] = (_Float16)a.y;
    h[2] = (_Float16)a.z; h[3] = (_Float16)a.w;
    h[4] = (_Float16)b.x; h[5] = (_Float16)b.y;
    h[6] = (_Float16)b.z; h[7] = (_Float16)b.w;
    *(v8h*)(xh + i) = h;
}

// w_q: int8 -> f16 with the per-(group, out_feature) scale folded in (f32 math,
// one f16 rounding). 16 consecutive k per thread => single group per thread.
__global__ __launch_bounds__(256) void cvt_w_scaled_f16(const int8_t* __restrict__ wq,
                                                        const float*  __restrict__ s_w,
                                                        _Float16*     __restrict__ wh) {
    const int i = (blockIdx.x * 256 + threadIdx.x) * 16;   // flat (o, k) index
    const int o = i >> 12;           // / 4096
    const int g = (i & 4095) >> 7;   // k / 128
    const float s = s_w[g * N_DIM + o];
    int4 p = *(const int4*)(wq + i);
    const int8_t* pb = (const int8_t*)&p;
    v16h h;
#pragma unroll
    for (int j = 0; j < 16; ++j) h[j] = (_Float16)((float)pb[j] * s);
    *(v16h*)(wh + i) = h;
}

// ---------------- WMMA GEMM ----------------
// Grid: (N/128, M/128). Block: 256 threads = 8 waves arranged 4(M) x 2(N).
// Each wave: 32 rows x 64 cols = 8 accumulators of 16x16; 8 independent WMMAs
// per 32-wide K step. Waves in a WG share A rows / B cols through WGP$.
__global__ __launch_bounds__(256, 1) void w4_gemm_wmma(
    const _Float16* __restrict__ xh,   // [M, K] row-major
    const _Float16* __restrict__ wh,   // [N, K] row-major, scales folded
    const float*    __restrict__ bias, // [N]
    float*          __restrict__ y)    // [M, N]
{
    const int lane = threadIdx.x & 31;
    const int wave = threadIdx.x >> 5;
    const int half = lane >> 4;     // 0: lanes 0-15, 1: lanes 16-31
    const int l16  = lane & 15;

    const int m0 = blockIdx.y * 128 + (wave >> 1) * 32;
    const int n0 = blockIdx.x * 128 + (wave & 1) * 64;

    // A: lane holds one row. Half 0 reads K 0..15, half 1 K 16..31.
    const _Float16* aptr0 = xh + (size_t)(m0 + l16) * K_DIM + (half ? 16 : 0);
    const _Float16* aptr1 = aptr0 + (size_t)16 * K_DIM;

    // B: lane holds one column; half 0 reads K 0..15, half 1 K 16..31 (contiguous 32B).
    const int col0 = n0 + 0  + l16;
    const int col1 = n0 + 16 + l16;
    const int col2 = n0 + 32 + l16;
    const int col3 = n0 + 48 + l16;
    const _Float16* bptr0 = wh + (size_t)col0 * K_DIM + (half ? 16 : 0);
    const _Float16* bptr1 = wh + (size_t)col1 * K_DIM + (half ? 16 : 0);
    const _Float16* bptr2 = wh + (size_t)col2 * K_DIM + (half ? 16 : 0);
    const _Float16* bptr3 = wh + (size_t)col3 * K_DIM + (half ? 16 : 0);

    v8f acc00 = {}, acc01 = {}, acc02 = {}, acc03 = {};
    v8f acc10 = {}, acc11 = {}, acc12 = {}, acc13 = {};

    for (int kc = 0; kc < K_DIM; kc += 128) {
        // WGP-scope prefetch two 128-K chunks (512 B) ahead to cover L2->L0 latency.
        prefetch_wgp(aptr0 + kc + 256);
        prefetch_wgp(aptr1 + kc + 256);
        prefetch_wgp(bptr0 + kc + 256);
        prefetch_wgp(bptr1 + kc + 256);
        prefetch_wgp(bptr2 + kc + 256);
        prefetch_wgp(bptr3 + kc + 256);
#pragma unroll
        for (int kk = 0; kk < 128; kk += 32) {
            const int k = kc + kk;
            // A fragments (two b128 loads each, combined in contiguous k order, as B)
            v8h a0lo = *(const v8h*)(aptr0 + k);
            v8h a0hi = *(const v8h*)(aptr0 + k + 8);
            v8h a1lo = *(const v8h*)(aptr1 + k);
            v8h a1hi = *(const v8h*)(aptr1 + k + 8);
            v16h a0, a1;
#pragma unroll
            for (int i = 0; i < 8; ++i) {
                a0[i] = a0lo[i]; a0[8 + i] = a0hi[i];
                a1[i] = a1lo[i]; a1[8 + i] = a1hi[i];
            }
            // B fragments (one 32B contiguous load each)
            v16h b0 = *(const v16h*)(bptr0 + k);
            v16h b1 = *(const v16h*)(bptr1 + k);
            v16h b2 = *(const v16h*)(bptr2 + k);
            v16h b3 = *(const v16h*)(bptr3 + k);

            acc00 = __builtin_amdgcn_wmma_f32_16x16x32_f16(false, a0, false, b0,
                                                           (short)0, acc00, false, false);
            acc01 = __builtin_amdgcn_wmma_f32_16x16x32_f16(false, a0, false, b1,
                                                           (short)0, acc01, false, false);
            acc02 = __builtin_amdgcn_wmma_f32_16x16x32_f16(false, a0, false, b2,
                                                           (short)0, acc02, false, false);
            acc03 = __builtin_amdgcn_wmma_f32_16x16x32_f16(false, a0, false, b3,
                                                           (short)0, acc03, false, false);
            acc10 = __builtin_amdgcn_wmma_f32_16x16x32_f16(false, a1, false, b0,
                                                           (short)0, acc10, false, false);
            acc11 = __builtin_amdgcn_wmma_f32_16x16x32_f16(false, a1, false, b1,
                                                           (short)0, acc11, false, false);
            acc12 = __builtin_amdgcn_wmma_f32_16x16x32_f16(false, a1, false, b2,
                                                           (short)0, acc12, false, false);
            acc13 = __builtin_amdgcn_wmma_f32_16x16x32_f16(false, a1, false, b3,
                                                           (short)0, acc13, false, false);
        }
    }

    // Epilogue. D layout: VGPR r -> row m0 + r + half*8 (m-frag 0) / +16 (m-frag 1);
    // lane -> column (consecutive lanes -> consecutive addresses: coalesced).
    const float b0v = bias[col0];
    const float b1v = bias[col1];
    const float b2v = bias[col2];
    const float b3v = bias[col3];
#pragma unroll
    for (int r = 0; r < 8; ++r) {
        const size_t row0 = (size_t)(m0 + r + half * 8);
        const size_t row1 = row0 + 16;
        y[row0 * N_DIM + col0] = acc00[r] + b0v;
        y[row0 * N_DIM + col1] = acc01[r] + b1v;
        y[row0 * N_DIM + col2] = acc02[r] + b2v;
        y[row0 * N_DIM + col3] = acc03[r] + b3v;
        y[row1 * N_DIM + col0] = acc10[r] + b0v;
        y[row1 * N_DIM + col1] = acc11[r] + b1v;
        y[row1 * N_DIM + col2] = acc12[r] + b2v;
        y[row1 * N_DIM + col3] = acc13[r] + b3v;
    }
}

// ---------------- launcher ----------------
extern "C" void kernel_launch(void* const* d_in, const int* in_sizes, int n_in,
                              void* d_out, int out_size, void* d_ws, size_t ws_size,
                              hipStream_t stream) {
    const float*  x    = (const float*)d_in[0];   // (2, 1024, 4096) f32
    const int8_t* w_q  = (const int8_t*)d_in[1];  // (4096, 32, 128) int8 == (N, K)
    const float*  s_w  = (const float*)d_in[2];   // (32, 4096) f32
    const float*  bias = (const float*)d_in[3];   // (4096,) f32
    float*        y    = (float*)d_out;           // (2048, 4096) f32

    // Workspace: xh (16 MB) then wh (32 MB)
    _Float16* xh = (_Float16*)d_ws;
    _Float16* wh = (_Float16*)((char*)d_ws + (size_t)M_DIM * K_DIM * sizeof(_Float16));

    cvt_x_f16<<<(M_DIM * K_DIM) / (256 * 8), 256, 0, stream>>>(x, xh);
    cvt_w_scaled_f16<<<(N_DIM * K_DIM) / (256 * 16), 256, 0, stream>>>(w_q, s_w, wh);

    dim3 grid(N_DIM / 128, M_DIM / 128);
    w4_gemm_wmma<<<grid, 256, 0, stream>>>(xh, wh, bias, y);
}
